// GPT_17008070492398
// MI455X (gfx1250) — compile-verified, matrix operand order checked by reading, not run on hardware
//
#include <hip/hip_runtime.h>
#include <hip/hip_bf16.h>

// ---------------------------------------------------------------------------
// GPT-2/MoE forward for MI455X (gfx1250, wave32, WMMA).
// All large GEMMs use v_wmma_f32_16x16x32_bf16 (f32 accumulate).
// bf16 x bf16 GEMM stages via GLOBAL_LOAD_ASYNC_TO_LDS (ASYNCcnt protocol).
// ---------------------------------------------------------------------------

typedef __attribute__((ext_vector_type(16))) __bf16 v16bf;
typedef __attribute__((ext_vector_type(8)))  __bf16 v8bf;
typedef __attribute__((ext_vector_type(8)))  float  v8f;
typedef __attribute__((ext_vector_type(4)))  float  v4f;
typedef __attribute__((ext_vector_type(4)))  int    v4i;

#define N_LAYER 2
#define N_HEAD  16
#define HD      64
#define C_EMB   1024
#define T_CTX   1024
#define NTOK    2048          // B*T
#define N_EXP   8
#define D_FFN   2048
#define TOT_W   16384         // N_EXP * D_FFN
#define VOCAB   50257
#define LDS_PAD 40            // padded BK row (conflict-free b128 frag loads)

#if __has_builtin(__builtin_amdgcn_global_load_async_to_lds_b128)
#define HAVE_ASYNC_LDS 1
// builtin signature: (int4 addrspace(1)* src, int4 addrspace(3)* dst, Ii off, Ii cpol)
#define GAS(p) ((__attribute__((address_space(1))) v4i*)(p))
#define LAS(p) ((__attribute__((address_space(3))) v4i*)(p))
#else
#define HAVE_ASYNC_LDS 0
#endif

#if __has_builtin(__builtin_amdgcn_s_wait_asynccnt)
#define WAIT_ASYNC() __builtin_amdgcn_s_wait_asynccnt(0)
#else
#define WAIT_ASYNC() asm volatile("s_wait_asynccnt 0" ::: "memory")
#endif

union FragAB { v16bf v; v8bf h[2]; };

__device__ __forceinline__ v8f vzero8() {
    v8f z;
#pragma unroll
    for (int i = 0; i < 8; ++i) z[i] = 0.0f;
    return z;
}
__device__ __forceinline__ v8bf bzero8() {
    v8bf z;
#pragma unroll
    for (int i = 0; i < 8; ++i) z[i] = (__bf16)0.0f;
    return z;
}

__device__ __forceinline__ v8f wmma_bf16(const FragAB& a, const FragAB& b, v8f c) {
    return __builtin_amdgcn_wmma_f32_16x16x32_bf16(false, a.v, false, b.v,
                                                   (short)0, c, false, false);
}

__device__ __forceinline__ float gelu_tanh(float x) {
    float x3 = x * x * x;
    return 0.5f * x * (1.0f + tanhf(0.7978845608028654f * (x + 0.044715f * x3)));
}

// load 8 contiguous elements as bf16 (convert if f32 source)
__device__ __forceinline__ v8bf loadcvt8(const float* p) {
    v4f a = *(const v4f*)p;
    v4f b = *(const v4f*)(p + 4);
    v8bf r;
#pragma unroll
    for (int i = 0; i < 4; ++i) r[i]     = (__bf16)a[i];
#pragma unroll
    for (int i = 0; i < 4; ++i) r[4 + i] = (__bf16)b[i];
    return r;
}
__device__ __forceinline__ v8bf loadcvt8(const __bf16* p) {
    return *(const v8bf*)p;
}

// ---------------------------------------------------------------------------
// Epilogues
// ---------------------------------------------------------------------------
enum { E_F32 = 0, E_RES = 1, E_GELU = 2, E_QKV = 3 };

template<int EPI>
__device__ __forceinline__ void epilogue(float acc, int m, int n, int ldo,
                                         float* outf, __bf16* outb,
                                         const float* residual, const float* gate,
                                         __bf16* qb, __bf16* kb, __bf16* vt) {
    if constexpr (EPI == E_F32) {
        outf[(size_t)m * ldo + n] = acc;
    } else if constexpr (EPI == E_RES) {
        size_t o = (size_t)m * ldo + n;
        outf[o] = residual[o] + acc;
    } else if constexpr (EPI == E_GELU) {
        float g = gate[m * N_EXP + (n >> 11)];           // expert = n / D_FFN
        outb[(size_t)m * ldo + n] = (__bf16)(gelu_tanh(acc) * g);
    } else {                                             // E_QKV scatter
        int sec = n >> 10;                               // 0=q 1=k 2=v
        int dg  = n & 1023;
        int h   = dg >> 6;
        int d   = dg & 63;
        int b   = m >> 10;
        int t   = m & 1023;
        size_t bh = (size_t)(b * N_HEAD + h) << 16;      // *T*HD = 65536
        __bf16 val = (__bf16)acc;
        if (sec == 0)      qb[bh + ((size_t)t << 6) + d] = val;   // [bh][t][d]
        else if (sec == 1) kb[bh + ((size_t)t << 6) + d] = val;   // [bh][t][d]
        else               vt[bh + ((size_t)d << 10) + t] = val;  // [bh][d][t]
    }
}

// shared fragment compute: 32x64 wave tile, 8 WMMAs per 32-deep K step
__device__ __forceinline__ void tile_mma(const __bf16 (*As)[LDS_PAD],
                                         const __bf16 (*Bs)[LDS_PAD],
                                         int wm, int wn, int fr, int cb,
                                         v8f c[2][4]) {
    FragAB a[2], b[4];
#pragma unroll
    for (int i = 0; i < 2; ++i) {
        int row = wm * 32 + i * 16 + fr;
        a[i].h[0] = *(const v8bf*)&As[row][cb];
        a[i].h[1] = *(const v8bf*)&As[row][cb + 16];
    }
#pragma unroll
    for (int j = 0; j < 4; ++j) {
        int row = wn * 64 + j * 16 + fr;
        b[j].h[0] = *(const v8bf*)&Bs[row][cb];
        b[j].h[1] = *(const v8bf*)&Bs[row][cb + 16];
    }
#pragma unroll
    for (int i = 0; i < 2; ++i)
#pragma unroll
        for (int j = 0; j < 4; ++j)
            c[i][j] = wmma_bf16(a[i], b[j], c[i][j]);
}

template<int EPI>
__device__ __forceinline__ void tile_epilogue(v8f c[2][4], int m0, int n0,
                                              int wm, int wn, int fr, int halfo,
                                              int M, int N, int ldo,
                                              float* outf, __bf16* outb,
                                              const float* residual, const float* gate,
                                              __bf16* qb, __bf16* kb, __bf16* vt) {
#pragma unroll
    for (int i = 0; i < 2; ++i)
#pragma unroll
        for (int j = 0; j < 4; ++j)
#pragma unroll
            for (int r = 0; r < 8; ++r) {
                int m = m0 + wm * 32 + i * 16 + r + halfo;
                int n = n0 + wn * 64 + j * 16 + fr;
                if (m < M && n < N)
                    epilogue<EPI>(c[i][j][r], m, n, ldo, outf, outb,
                                  residual, gate, qb, kb, vt);
            }
}

// ---------------------------------------------------------------------------
// Generic NT GEMM:  C[M,N] = A[M,K] * B[N,K]^T ; bf16 WMMA, f32 accum.
// Block tile 128x128, 8 waves (4x2) of 32x64, BK=32.
// Register-level pipeline: next K tile fetched while current LDS tile computes.
// ---------------------------------------------------------------------------
template<typename TA, typename TB, int EPI>
__global__ __launch_bounds__(256) void gemm_nt(
    const TA* __restrict__ A, const TB* __restrict__ B,
    int M, int N, int K,
    float* __restrict__ outf, __bf16* __restrict__ outb, int ldo,
    const float* __restrict__ residual, const float* __restrict__ gate,
    __bf16* __restrict__ qb, __bf16* __restrict__ kb, __bf16* __restrict__ vt)
{
    __shared__ __bf16 As[128][LDS_PAD];
    __shared__ __bf16 Bs[128][LDS_PAD];

    const int tid  = threadIdx.x;
    const int lane = tid & 31;
    const int wid  = tid >> 5;
    const int wm   = wid >> 1;          // 0..3
    const int wn   = wid & 1;           // 0..1
    const int m0   = blockIdx.y * 128;
    const int n0   = blockIdx.x * 128;

    const int fr    = lane & 15;               // frag row within 16
    const int cb    = (lane < 16) ? 0 : 8;     // K-chunk base per half-wave
    const int halfo = (lane >> 4) * 8;

    // staging: 128 rows x (2 threads * 16 cols)
    const int r128 = tid >> 1;
    const int c16  = (tid & 1) * 16;

    v8f c[2][4];
#pragma unroll
    for (int i = 0; i < 2; ++i)
#pragma unroll
        for (int j = 0; j < 4; ++j) c[i][j] = vzero8();

    v8bf ra0, ra1, rb0, rb1;
    auto fetch = [&](int k0) {
        int gm = m0 + r128;
        if (gm < M) {
            const TA* pa = A + (size_t)gm * K + k0 + c16;
            ra0 = loadcvt8(pa); ra1 = loadcvt8(pa + 8);
        } else { ra0 = bzero8(); ra1 = bzero8(); }
        int gn = n0 + r128;
        if (gn < N) {
            const TB* pb = B + (size_t)gn * K + k0 + c16;
            rb0 = loadcvt8(pb); rb1 = loadcvt8(pb + 8);
        } else { rb0 = bzero8(); rb1 = bzero8(); }
    };

    fetch(0);
    for (int k0 = 0; k0 < K; k0 += 32) {
        // commit staged registers to LDS
        *(v8bf*)&As[r128][c16]     = ra0;
        *(v8bf*)&As[r128][c16 + 8] = ra1;
        *(v8bf*)&Bs[r128][c16]     = rb0;
        *(v8bf*)&Bs[r128][c16 + 8] = rb1;
        __syncthreads();
        // overlap: fetch next tile + prefetch one ahead (global_prefetch_b8)
        if (k0 + 32 < K) fetch(k0 + 32);
        if (k0 + 64 < K) {
            if (m0 + r128 < M) __builtin_prefetch(A + (size_t)(m0 + r128) * K + k0 + 64 + c16, 0, 1);
            if (n0 + r128 < N) __builtin_prefetch(B + (size_t)(n0 + r128) * K + k0 + 64 + c16, 0, 1);
        }
        tile_mma(As, Bs, wm, wn, fr, cb, c);
        __syncthreads();
    }

    tile_epilogue<EPI>(c, m0, n0, wm, wn, fr, halfo, M, N, ldo,
                       outf, outb, residual, gate, qb, kb, vt);
}

// ---------------------------------------------------------------------------
// bf16 x bf16 NT GEMM with async global->LDS staging (ASYNCcnt protocol).
// Requires M,N multiples of 128 and K multiple of 32 (true for MoE GEMM2).
// ---------------------------------------------------------------------------
#if HAVE_ASYNC_LDS
template<int EPI>
__global__ __launch_bounds__(256) void gemm_nt_async(
    const __bf16* __restrict__ A, const __bf16* __restrict__ B,
    int M, int N, int K,
    float* __restrict__ outf, __bf16* __restrict__ outb, int ldo,
    const float* __restrict__ residual, const float* __restrict__ gate,
    __bf16* __restrict__ qb, __bf16* __restrict__ kb, __bf16* __restrict__ vt)
{
    __shared__ __bf16 As[2][128][LDS_PAD];
    __shared__ __bf16 Bs[2][128][LDS_PAD];

    const int tid  = threadIdx.x;
    const int lane = tid & 31;
    const int wid  = tid >> 5;
    const int wm   = wid >> 1;
    const int wn   = wid & 1;
    const int m0   = blockIdx.y * 128;
    const int n0   = blockIdx.x * 128;

    const int fr    = lane & 15;
    const int cb    = (lane < 16) ? 0 : 8;
    const int halfo = (lane >> 4) * 8;
    const int r128  = tid >> 1;
    const int c16   = (tid & 1) * 16;

    v8f c[2][4];
#pragma unroll
    for (int i = 0; i < 2; ++i)
#pragma unroll
        for (int j = 0; j < 4; ++j) c[i][j] = vzero8();

    auto issue = [&](int buf, int k0) {
        const __bf16* pa = A + (size_t)(m0 + r128) * K + k0 + c16;
        const __bf16* pb = B + (size_t)(n0 + r128) * K + k0 + c16;
        __builtin_amdgcn_global_load_async_to_lds_b128(
            GAS(pa),     LAS(&As[buf][r128][c16]),     0, 0);
        __builtin_amdgcn_global_load_async_to_lds_b128(
            GAS(pa + 8), LAS(&As[buf][r128][c16 + 8]), 0, 0);
        __builtin_amdgcn_global_load_async_to_lds_b128(
            GAS(pb),     LAS(&Bs[buf][r128][c16]),     0, 0);
        __builtin_amdgcn_global_load_async_to_lds_b128(
            GAS(pb + 8), LAS(&Bs[buf][r128][c16 + 8]), 0, 0);
    };

    int buf = 0;
    issue(0, 0);
    for (int k0 = 0; k0 < K; k0 += 32) {
        WAIT_ASYNC();            // own async writes to LDS complete
        __syncthreads();         // all waves' tiles visible; prev reads done
        if (k0 + 32 < K) issue(buf ^ 1, k0 + 32);
        tile_mma(As[buf], Bs[buf], wm, wn, fr, cb, c);
        buf ^= 1;
    }

    tile_epilogue<EPI>(c, m0, n0, wm, wn, fr, halfo, M, N, ldo,
                       outf, outb, residual, gate, qb, kb, vt);
}
#endif

// ---------------------------------------------------------------------------
// Flash attention: one wave per (b,h, 16 q-rows). Causal, online softmax.
// q,k: [bh][t][64] bf16 ; vT: [bh][64][t] bf16 ; y out f32 [n,1024].
// ---------------------------------------------------------------------------
__global__ __launch_bounds__(32) void attn_kernel(
    const __bf16* __restrict__ qb, const __bf16* __restrict__ kb,
    const __bf16* __restrict__ vt, float* __restrict__ y)
{
    __shared__ __bf16 plds[16][LDS_PAD];

    const int lane  = threadIdx.x;
    const int q0    = blockIdx.x * 16;
    const int bh    = blockIdx.y;
    const int b     = bh >> 4;
    const int h     = bh & 15;
    const int fr    = lane & 15;
    const int cbk   = (lane < 16) ? 0 : 8;
    const int halfo = (lane >> 4) * 8;
    const size_t base = (size_t)bh << 16;      // *T*HD

    FragAB aq[2];
#pragma unroll
    for (int s = 0; s < 2; ++s) {
        const __bf16* qr = qb + base + (size_t)(q0 + fr) * HD + s * 32;
        aq[s].h[0] = *(const v8bf*)(qr + cbk);
        aq[s].h[1] = *(const v8bf*)(qr + cbk + 16);
    }

    v8f o[4];
#pragma unroll
    for (int nt = 0; nt < 4; ++nt) o[nt] = vzero8();
    float mrow[8], lrow[8];
#pragma unroll
    for (int r = 0; r < 8; ++r) { mrow[r] = -3.0e38f; lrow[r] = 0.0f; }

    const int nkb = (q0 + 16 + 31) >> 5;       // causal bound, key blocks of 32
    for (int tkb = 0; tkb < nkb; ++tkb) {
        const int kpos0 = tkb * 32;
        v8f s[2];
        s[0] = vzero8(); s[1] = vzero8();

        // scores = q @ k^T  (two 16x16 n-tiles)
#pragma unroll
        for (int t = 0; t < 2; ++t) {
            const __bf16* kr = kb + base + (size_t)(kpos0 + t * 16 + fr) * HD;
            FragAB bk0, bk1;
            bk0.h[0] = *(const v8bf*)(kr + cbk);
            bk0.h[1] = *(const v8bf*)(kr + cbk + 16);
            bk1.h[0] = *(const v8bf*)(kr + 32 + cbk);
            bk1.h[1] = *(const v8bf*)(kr + 32 + cbk + 16);
            s[t] = wmma_bf16(aq[0], bk0, s[t]);
            s[t] = wmma_bf16(aq[1], bk1, s[t]);
        }

        // scale + causal mask; rows align with accumulator VGPR index
        float mloc[8];
#pragma unroll
        for (int r = 0; r < 8; ++r) {
            int row = q0 + r + halfo;
#pragma unroll
            for (int t = 0; t < 2; ++t) {
                int col = kpos0 + t * 16 + fr;
                float v = s[t][r] * 0.125f;                 // 1/sqrt(64)
                s[t][r] = (col <= row) ? v : -3.0e38f;
            }
            mloc[r] = fmaxf(s[0][r], s[1][r]);
        }
#pragma unroll
        for (int off = 8; off >= 1; off >>= 1)
#pragma unroll
            for (int r = 0; r < 8; ++r)
                mloc[r] = fmaxf(mloc[r], __shfl_xor(mloc[r], off, 16));

        float corr[8], rs[8];
#pragma unroll
        for (int r = 0; r < 8; ++r) {
            float mn = fmaxf(mrow[r], mloc[r]);
            corr[r] = __expf(mrow[r] - mn);
            mrow[r] = mn;
            s[0][r] = __expf(s[0][r] - mn);
            s[1][r] = __expf(s[1][r] - mn);
            rs[r] = s[0][r] + s[1][r];
        }
#pragma unroll
        for (int off = 8; off >= 1; off >>= 1)
#pragma unroll
            for (int r = 0; r < 8; ++r)
                rs[r] += __shfl_xor(rs[r], off, 16);
#pragma unroll
        for (int r = 0; r < 8; ++r) lrow[r] = lrow[r] * corr[r] + rs[r];
#pragma unroll
        for (int nt = 0; nt < 4; ++nt)
#pragma unroll
            for (int r = 0; r < 8; ++r) o[nt][r] *= corr[r];

        // C-frag -> A-frag relayout of p through LDS (single-wave WG sync)
        __syncthreads();
#pragma unroll
        for (int t = 0; t < 2; ++t)
#pragma unroll
            for (int r = 0; r < 8; ++r)
                plds[r + halfo][t * 16 + fr] = (__bf16)s[t][r];
        __syncthreads();
        FragAB p;
        p.h[0] = *(const v8bf*)&plds[fr][cbk];
        p.h[1] = *(const v8bf*)&plds[fr][cbk + 16];
        __syncthreads();

        // o += p @ v  (vT rows are hd-major -> NT fragment loads)
#pragma unroll
        for (int nt = 0; nt < 4; ++nt) {
            const __bf16* vr = vt + base + (size_t)(nt * 16 + fr) * T_CTX + kpos0;
            FragAB bv;
            bv.h[0] = *(const v8bf*)(vr + cbk);
            bv.h[1] = *(const v8bf*)(vr + cbk + 16);
            o[nt] = wmma_bf16(p, bv, o[nt]);
        }
    }

    // normalize and scatter to y[n, 1024]
#pragma unroll
    for (int r = 0; r < 8; ++r) {
        float inv = 1.0f / lrow[r];
        int row = q0 + r + halfo;
#pragma unroll
        for (int nt = 0; nt < 4; ++nt) {
            int col = h * HD + nt * 16 + fr;
            y[((size_t)b * T_CTX + row) * C_EMB + col] = o[nt][r] * inv;
        }
    }
}

// ---------------------------------------------------------------------------
// Embedding: x[n] = wte[idx[n]] + wpe[n % T]
// ---------------------------------------------------------------------------
__global__ __launch_bounds__(256) void embed_kernel(
    const int* __restrict__ idx, const float* __restrict__ wte,
    const float* __restrict__ wpe, float* __restrict__ x)
{
    int n = blockIdx.x;
    int t = n & (T_CTX - 1);
    int c = threadIdx.x * 4;
    int tok = idx[n];
    v4f a = *(const v4f*)(wte + (size_t)tok * C_EMB + c);
    v4f b = *(const v4f*)(wpe + (size_t)t * C_EMB + c);
    *(v4f*)(x + (size_t)n * C_EMB + c) = a + b;
}

// ---------------------------------------------------------------------------
// LayerNorm (bias-free, eps 1e-5): one wave per row of 1024.
// ---------------------------------------------------------------------------
__global__ __launch_bounds__(256) void ln_kernel(
    const float* __restrict__ x, const float* __restrict__ w,
    float* __restrict__ out)
{
    int lane = threadIdx.x & 31;
    int wid  = threadIdx.x >> 5;
    int row  = blockIdx.x * 8 + wid;
    const float* xr = x + (size_t)row * C_EMB;

    float v[32], s = 0.0f;
#pragma unroll
    for (int i = 0; i < 32; ++i) { v[i] = xr[lane + i * 32]; s += v[i]; }
#pragma unroll
    for (int off = 16; off >= 1; off >>= 1) s += __shfl_xor(s, off, 32);
    float mu = s * (1.0f / C_EMB);

    float vs = 0.0f;
#pragma unroll
    for (int i = 0; i < 32; ++i) { float d = v[i] - mu; vs += d * d; }
#pragma unroll
    for (int off = 16; off >= 1; off >>= 1) vs += __shfl_xor(vs, off, 32);
    float rinv = rsqrtf(vs * (1.0f / C_EMB) + 1e-5f);

    float* orow = out + (size_t)row * C_EMB;
#pragma unroll
    for (int i = 0; i < 32; ++i) {
        int c = lane + i * 32;
        orow[c] = (v[i] - mu) * rinv * w[c];
    }
}

// ---------------------------------------------------------------------------
// Router: one wave per token. softmax -> top2 -> renormalize -> dense gate[8].
// ---------------------------------------------------------------------------
__global__ __launch_bounds__(32) void router_kernel(
    const float* __restrict__ xn, const float* __restrict__ rw,
    float* __restrict__ gate)
{
    int n = blockIdx.x;
    int lane = threadIdx.x;
    const float* xr = xn + (size_t)n * C_EMB;

    float acc[N_EXP];
#pragma unroll
    for (int e = 0; e < N_EXP; ++e) acc[e] = 0.0f;
    for (int i = 0; i < 32; ++i) {
        float xv = xr[lane + i * 32];
#pragma unroll
        for (int e = 0; e < N_EXP; ++e)
            acc[e] += xv * rw[e * C_EMB + lane + i * 32];
    }
#pragma unroll
    for (int off = 16; off >= 1; off >>= 1)
#pragma unroll
        for (int e = 0; e < N_EXP; ++e)
            acc[e] += __shfl_xor(acc[e], off, 32);

    float mx = acc[0];
#pragma unroll
    for (int e = 1; e < N_EXP; ++e) mx = fmaxf(mx, acc[e]);
    float p[N_EXP], ps = 0.0f;
#pragma unroll
    for (int e = 0; e < N_EXP; ++e) { p[e] = __expf(acc[e] - mx); ps += p[e]; }
#pragma unroll
    for (int e = 0; e < N_EXP; ++e) p[e] /= ps;

    int i1 = 0;
#pragma unroll
    for (int e = 1; e < N_EXP; ++e) if (p[e] > p[i1]) i1 = e;   // lowest idx on tie
    int i2 = (i1 == 0) ? 1 : 0;
#pragma unroll
    for (int e = 0; e < N_EXP; ++e) if (e != i1 && p[e] > p[i2]) i2 = e;
    float s2 = p[i1] + p[i2];

    if (lane < N_EXP)
        gate[(size_t)n * N_EXP + lane] =
            (lane == i1) ? p[i1] / s2 : ((lane == i2) ? p[i2] / s2 : 0.0f);
}

// ---------------------------------------------------------------------------
// Transpose + convert: in f32 [R][Cc] -> out bf16 [Cc][R]  (32x32 tiles)
// ---------------------------------------------------------------------------
__global__ __launch_bounds__(256) void transpose_cvt(
    const float* __restrict__ in, __bf16* __restrict__ out, int R, int Cc)
{
    __shared__ float tile[32][33];
    int tx = threadIdx.x & 31, ty = threadIdx.x >> 5;
    int c0 = blockIdx.x * 32, r0 = blockIdx.y * 32;
#pragma unroll
    for (int j = 0; j < 4; ++j)
        tile[ty + j * 8][tx] = in[(size_t)(r0 + ty + j * 8) * Cc + c0 + tx];
    __syncthreads();
#pragma unroll
    for (int j = 0; j < 4; ++j)
        out[(size_t)(c0 + ty + j * 8) * R + r0 + tx] = (__bf16)tile[tx][ty + j * 8];
}

// ---------------------------------------------------------------------------
// Host launch
// ---------------------------------------------------------------------------
template<typename TA, typename TB, int EPI>
static void launch_gemm(hipStream_t s, const TA* A, const TB* B, int M, int N, int K,
                        float* outf, __bf16* outb, int ldo,
                        const float* res, const float* gate,
                        __bf16* qb, __bf16* kb, __bf16* vt) {
    dim3 grid((N + 127) / 128, (M + 127) / 128);
    gemm_nt<TA, TB, EPI><<<grid, 256, 0, s>>>(A, B, M, N, K, outf, outb, ldo,
                                              res, gate, qb, kb, vt);
}

extern "C" void kernel_launch(void* const* d_in, const int* in_sizes, int n_in,
                              void* d_out, int out_size, void* d_ws, size_t ws_size,
                              hipStream_t stream) {
    (void)in_sizes; (void)n_in; (void)out_size; (void)ws_size;
    const int*   idx      = (const int*)d_in[0];
    const float* wte      = (const float*)d_in[1];
    const float* wpe      = (const float*)d_in[2];
    const float* ln1_w    = (const float*)d_in[3];
    const float* qkv_w    = (const float*)d_in[4];
    const float* proj_w   = (const float*)d_in[5];
    const float* ln2_w    = (const float*)d_in[6];
    const float* router_w = (const float*)d_in[7];
    const float* w1       = (const float*)d_in[8];
    const float* w2       = (const float*)d_in[9];
    const float* lnf_w    = (const float*)d_in[10];
    float* out = (float*)d_out;

    // ---- carve workspace (~164 MB) ----
    char* ws = (char*)d_ws;
    auto carve = [&](size_t bytes) -> void* {
        void* p = (void*)ws;
        ws += (bytes + 255) & ~(size_t)255;
        return p;
    };
    float*  x    = (float*)carve((size_t)NTOK * C_EMB * 4);
    float*  xn   = (float*)carve((size_t)NTOK * C_EMB * 4);
    float*  y    = (float*)carve((size_t)NTOK * C_EMB * 4);
    __bf16* qb   = (__bf16*)carve((size_t)NTOK * C_EMB * 2);
    __bf16* kbuf = (__bf16*)carve((size_t)NTOK * C_EMB * 2);
    __bf16* vt   = (__bf16*)carve((size_t)NTOK * C_EMB * 2);
    float*  gate = (float*)carve((size_t)NTOK * N_EXP * 4);
    __bf16* w1t  = (__bf16*)carve((size_t)TOT_W * C_EMB * 2);
    __bf16* w2t  = (__bf16*)carve((size_t)TOT_W * C_EMB * 2);
    __bf16* Hg   = (__bf16*)carve((size_t)NTOK * TOT_W * 2);

    embed_kernel<<<NTOK, 256, 0, stream>>>(idx, wte, wpe, x);

    for (int l = 0; l < N_LAYER; ++l) {
        const float* qw  = qkv_w    + (size_t)l * 3 * C_EMB * C_EMB;
        const float* pw  = proj_w   + (size_t)l * C_EMB * C_EMB;
        const float* rw  = router_w + (size_t)l * N_EXP * C_EMB;
        const float* w1l = w1 + (size_t)l * C_EMB * TOT_W;
        const float* w2l = w2 + (size_t)l * TOT_W * C_EMB;

        // ln1 -> xn ; qkv = xn @ qkv_w^T (scatter bf16 q/k/vT per head)
        ln_kernel<<<NTOK / 8, 256, 0, stream>>>(x, ln1_w + l * C_EMB, xn);
        launch_gemm<float, float, E_QKV>(stream, xn, qw, NTOK, 3 * C_EMB, C_EMB,
                                         nullptr, nullptr, 0, nullptr, nullptr,
                                         qb, kbuf, vt);

        // flash attention -> y
        attn_kernel<<<dim3(T_CTX / 16, 2 * N_HEAD), 32, 0, stream>>>(qb, kbuf, vt, y);

        // x += y @ proj_w^T
        launch_gemm<float, float, E_RES>(stream, y, pw, NTOK, C_EMB, C_EMB,
                                         x, nullptr, C_EMB, x, nullptr,
                                         nullptr, nullptr, nullptr);

        // ln2 -> xn ; router gates
        ln_kernel<<<NTOK / 8, 256, 0, stream>>>(x, ln2_w + l * C_EMB, xn);
        router_kernel<<<NTOK, 32, 0, stream>>>(xn, rw, gate);

        // MoE: Hg = gelu(xn @ w1) * gate ; x += Hg @ w2
        transpose_cvt<<<dim3(TOT_W / 32, C_EMB / 32), 256, 0, stream>>>(w1l, w1t, C_EMB, TOT_W);
        launch_gemm<float, __bf16, E_GELU>(stream, xn, w1t, NTOK, TOT_W, C_EMB,
                                           nullptr, Hg, TOT_W, nullptr, gate,
                                           nullptr, nullptr, nullptr);
        transpose_cvt<<<dim3(C_EMB / 32, TOT_W / 32), 256, 0, stream>>>(w2l, w2t, TOT_W, C_EMB);
#if HAVE_ASYNC_LDS
        {
            dim3 grid(C_EMB / 128, NTOK / 128);
            gemm_nt_async<E_RES><<<grid, 256, 0, stream>>>(
                Hg, w2t, NTOK, C_EMB, TOT_W, x, nullptr, C_EMB, x, nullptr,
                nullptr, nullptr, nullptr);
        }
#else
        launch_gemm<__bf16, __bf16, E_RES>(stream, Hg, w2t, NTOK, C_EMB, TOT_W,
                                           x, nullptr, C_EMB, x, nullptr,
                                           nullptr, nullptr, nullptr);
#endif
    }

    // final LN + tied lm_head: out = ln(x, lnf) @ wte^T
    ln_kernel<<<NTOK / 8, 256, 0, stream>>>(x, lnf_w, xn);
    launch_gemm<float, float, E_F32>(stream, xn, wte, NTOK, VOCAB, C_EMB,
                                     out, nullptr, VOCAB, nullptr, nullptr,
                                     nullptr, nullptr, nullptr);
}
